// AttentionBlock_46076409151820
// MI455X (gfx1250) — matrix operand order repaired and checked by reading, not compiled, not run on hardware
//
#include <hip/hip_runtime.h>
#include <hip/hip_bf16.h>

// ---------------------------------------------------------------------------
// AttentionBlock for MI455X (gfx1250, wave32, WMMA).
//   B=4, C=64, H=W=64  ->  N=4096 tokens/batch, BN=16384 tokens total.
// All five 64x64 GEMMs (q,k,v,FFN1,FFN2) and both attention GEMMs run on
// v_wmma_f32_16x16x32_f16. Attention is flash-style (S^T formulation keeps
// one softmax row per lane; no LDS transpose for P), with double-buffered
// K/V tiles staged by GLOBAL_LOAD_ASYNC_TO_LDS_B128 (ASYNCcnt-tracked DMA).
// ---------------------------------------------------------------------------

#define CCH   64
#define NTOK  4096
#define BN    16384
#define WSTR  72   // LDS row stride (halves) for weight matrices

// Flash-tile LDS strides: row byte offset must be 16B-aligned for B128 LDS
// writes; stride in banks must have gcd(.,64) <= 4 for conflict-free 16-lane
// fragment reads.  88 halves = 176B (44 banks), 56 halves = 112B (28 banks).
#define KSTR  88
#define VSTR  56

#if defined(__has_builtin)
#if __has_builtin(__builtin_amdgcn_global_load_async_to_lds_b128) && \
    __has_builtin(__builtin_amdgcn_s_wait_asynccnt)
#define USE_ASYNC_LDS 1
#endif
#endif
#ifndef USE_ASYNC_LDS
#define USE_ASYNC_LDS 0
#endif

typedef __attribute__((ext_vector_type(16))) _Float16 v16h;
typedef __attribute__((ext_vector_type(8)))  float    v8f;
typedef __attribute__((ext_vector_type(8)))  unsigned v8u;

#if USE_ASYNC_LDS
typedef int v4i __attribute__((vector_size(16)));
typedef __attribute__((address_space(1))) v4i* glds_gptr;   // global int4*
typedef __attribute__((address_space(3))) v4i* glds_lptr;   // LDS    int4*
#endif

__device__ __forceinline__ v8f wmma_f16(v8u a, v8u b, v8f c) {
  return __builtin_amdgcn_wmma_f32_16x16x32_f16(
      /*neg_a=*/false, __builtin_bit_cast(v16h, a),
      /*neg_b=*/false, __builtin_bit_cast(v16h, b),
      /*c_mod=*/(short)0, c, /*reuse_a=*/false, /*reuse_b=*/false);
}

__device__ __forceinline__ unsigned pk16(float a, float b) {
  unsigned short lo = __builtin_bit_cast(unsigned short, (_Float16)a);
  unsigned short hi = __builtin_bit_cast(unsigned short, (_Float16)b);
  return (unsigned)lo | ((unsigned)hi << 16);
}

// A-fragment load from an LDS-resident 64x64 f16 matrix (row stride WSTR):
// rows = chunk*16 + l16 block, K-slice = khalf*32 .. +31.
__device__ __forceinline__ v8u ldA(const unsigned* w32, int chunk, int khalf,
                                   int l16, int hf) {
  v8u a;
#pragma unroll
  for (int j = 0; j < 8; ++j) {
    a[j] = w32[(chunk * 16 + l16) * (WSTR / 2) + khalf * 16 +
               hf * 8 + j];
  }
  return a;
}

// ---------------------------------------------------------------------------
// Kernel 1: WMMA projections. Wave handles 16 tokens.
// ---------------------------------------------------------------------------
__global__ __launch_bounds__(256) void proj_kernel(
    const float* __restrict__ seg, const float* __restrict__ gau,
    const float* __restrict__ Wq, const float* __restrict__ bq,
    const float* __restrict__ Wk, const float* __restrict__ bk,
    const float* __restrict__ Wv, const float* __restrict__ bv,
    _Float16* __restrict__ qf, _Float16* __restrict__ kf,
    float* __restrict__ vf, _Float16* __restrict__ vtf) {
  __shared__ _Float16 Wl[3][CCH * WSTR];
  {
    const float* Ws[3] = {Wq, Wk, Wv};
#pragma unroll
    for (int m = 0; m < 3; ++m) {
      for (int e = threadIdx.x; e < CCH * CCH; e += 256) {
        const int o = e >> 6, c = e & 63;
        Wl[m][o * WSTR + c] = (_Float16)Ws[m][e];
      }
    }
  }
  __syncthreads();

  const int lane = threadIdx.x & 31;
  const int wave = threadIdx.x >> 5;
  const int hf = lane >> 4, l16 = lane & 15;
  const int tok0 = blockIdx.x * 128 + wave * 16;
  const int b = tok0 >> 12;
  const int tok = tok0 + l16;
  const int n = tok & (NTOK - 1);

  const float* sp = seg + (b * CCH) * NTOK + n;
  const float* gp = gau + (b * CCH) * NTOK + n;
  v8u bsl, bsh, bgl, bgh;
#pragma unroll
  for (int j = 0; j < 8; ++j) {
    const int c0 = hf * 16 + 2 * j;
    bsl[j] = pk16(sp[c0 * NTOK], sp[(c0 + 1) * NTOK]);
    bsh[j] = pk16(sp[(c0 + 32) * NTOK], sp[(c0 + 33) * NTOK]);
    bgl[j] = pk16(gp[c0 * NTOK], gp[(c0 + 1) * NTOK]);
    bgh[j] = pk16(gp[(c0 + 32) * NTOK], gp[(c0 + 33) * NTOK]);
  }

  const unsigned* wq32 = (const unsigned*)&Wl[0][0];
  const unsigned* wk32 = (const unsigned*)&Wl[1][0];
  const unsigned* wv32 = (const unsigned*)&Wl[2][0];

  unsigned* qo32 = (unsigned*)(qf + tok * CCH);
  unsigned* ko32 = (unsigned*)(kf + tok * CCH);
  float* vo = vf + tok * CCH;

#pragma unroll
  for (int chunk = 0; chunk < 4; ++chunk) {
    v8f dq = {};
    dq = wmma_f16(ldA(wq32, chunk, 0, l16, hf), bsl, dq);
    dq = wmma_f16(ldA(wq32, chunk, 1, l16, hf), bsh, dq);
    v8f dk = {};
    dk = wmma_f16(ldA(wk32, chunk, 0, l16, hf), bsl, dk);
    dk = wmma_f16(ldA(wk32, chunk, 1, l16, hf), bsh, dk);
    v8f dv = {};
    dv = wmma_f16(ldA(wv32, chunk, 0, l16, hf), bgl, dv);
    dv = wmma_f16(ldA(wv32, chunk, 1, l16, hf), bgh, dv);

#pragma unroll
    for (int i = 0; i < 4; ++i) {
      const int ch = chunk * 16 + 8 * hf + 2 * i;
      qo32[chunk * 8 + hf * 4 + i] = pk16(dq[2 * i] + bq[ch], dq[2 * i + 1] + bq[ch + 1]);
      ko32[chunk * 8 + hf * 4 + i] = pk16(dk[2 * i] + bk[ch], dk[2 * i + 1] + bk[ch + 1]);
    }
#pragma unroll
    for (int r = 0; r < 8; ++r) {
      const int ch = chunk * 16 + 8 * hf + r;
      const float val = dv[r] + bv[ch];
      vo[ch] = val;
      vtf[(b * CCH + ch) * NTOK + n] = (_Float16)val;
    }
  }
}

// ---------------------------------------------------------------------------
// Kernel 2: flash attention; double-buffered LDS tiles, async DMA if present.
// ---------------------------------------------------------------------------
__global__ __launch_bounds__(256) void flash_attn_kernel(
    const _Float16* __restrict__ qf, const _Float16* __restrict__ kf,
    const _Float16* __restrict__ vtf, float* __restrict__ of) {
  __shared__ _Float16 Klds[2 * 32 * KSTR];
  __shared__ _Float16 Vlds[2 * 64 * VSTR];
  unsigned* klds32 = (unsigned*)Klds;
  unsigned* vlds32 = (unsigned*)Vlds;

  const int lane = threadIdx.x & 31;
  const int wave = threadIdx.x >> 5;
  const int hf = lane >> 4, l16 = lane & 15;

  const int b = blockIdx.x >> 5;
  const int qbase = ((int)blockIdx.x & 31) * 128 + wave * 16;
  const int tokQ = b * NTOK + qbase;

  const unsigned* qp = (const unsigned*)(qf + (tokQ + l16) * CCH);
  v8u bql, bqh;
#pragma unroll
  for (int j = 0; j < 8; ++j) {
    bql[j] = qp[(j >> 2) * 8 + hf * 4 + (j & 3)];
    bqh[j] = qp[16 + (j >> 2) * 8 + hf * 4 + (j & 3)];
  }

  v8f acc0 = {}, acc1 = {}, acc2 = {}, acc3 = {};
  float M = -1e30f, L = 0.0f;
  const float cs = 0.125f * 1.44269504f;   // 1/sqrt(C) * log2(e)

  const unsigned* kg = (const unsigned*)(kf + (b * NTOK) * CCH);
  const unsigned* vg = (const unsigned*)(vtf + (b * CCH) * NTOK);

  const int kRow = threadIdx.x >> 3, kPart = ((int)threadIdx.x & 7) * 4;
  const int vRow = threadIdx.x >> 2, vPart = ((int)threadIdx.x & 3) * 4;

  // Stage one 32-key tile (16B per thread per matrix) into buffer `buf`.
  auto stage = [&](int kt, int buf) {
    const unsigned* src  = kg + (kt * 32 + kRow) * 32 + kPart;
    const unsigned* vsrc = vg + vRow * (NTOK / 2) + kt * 16 + vPart;
    unsigned* kdst = &klds32[buf * (32 * KSTR / 2) + kRow * (KSTR / 2) + kPart];
    unsigned* vdst = &vlds32[buf * (64 * VSTR / 2) + vRow * (VSTR / 2) + vPart];
#if USE_ASYNC_LDS
    // Per-lane 16B global->LDS DMA, tracked by ASYNCcnt (no VGPR round-trip).
    __builtin_amdgcn_global_load_async_to_lds_b128(
        (glds_gptr)(unsigned long long)src,
        (glds_lptr)(unsigned)(unsigned long long)kdst, 0, 0);
    __builtin_amdgcn_global_load_async_to_lds_b128(
        (glds_gptr)(unsigned long long)vsrc,
        (glds_lptr)(unsigned)(unsigned long long)vdst, 0, 0);
#else
#pragma unroll
    for (int i = 0; i < 4; ++i) kdst[i] = src[i];
#pragma unroll
    for (int i = 0; i < 4; ++i) vdst[i] = vsrc[i];
#endif
  };

  stage(0, 0);   // prologue

  for (int kt = 0; kt < NTOK / 32; ++kt) {
    const int cur = kt & 1;
#if USE_ASYNC_LDS
    __builtin_amdgcn_s_wait_asynccnt(0);
#endif
    __syncthreads();   // tile `cur` resident; previous compute finished
    if (kt + 1 < NTOK / 32) {
      stage(kt + 1, cur ^ 1);   // overlap next copy with this tile's WMMAs
      __builtin_prefetch(kg + ((kt + 1) * 32 + kRow) * 32 + kPart, 0, 0);
    }
    const int kb = cur * (32 * KSTR / 2);
    const int vb = cur * (64 * VSTR / 2);

    // ---- S^T = K * Q^T for two 16-key groups ----
    v8f s0, s1;
    {
      v8u a0, a1;
#pragma unroll
      for (int j = 0; j < 8; ++j) {
        const int idx = kb + l16 * (KSTR / 2) + (j >> 2) * 8 + hf * 4 + (j & 3);
        a0[j] = klds32[idx];
        a1[j] = klds32[idx + 16];
      }
      v8f c0 = {};
      c0 = wmma_f16(a0, bql, c0);
      s0 = wmma_f16(a1, bqh, c0);
#pragma unroll
      for (int j = 0; j < 8; ++j) {
        const int idx = kb + (16 + l16) * (KSTR / 2) + (j >> 2) * 8 + hf * 4 + (j & 3);
        a0[j] = klds32[idx];
        a1[j] = klds32[idx + 16];
      }
      v8f c1 = {};
      c1 = wmma_f16(a0, bql, c1);
      s1 = wmma_f16(a1, bqh, c1);
    }

    // ---- online softmax: one q-row per lane ----
    float mt = -1e30f;
#pragma unroll
    for (int r = 0; r < 8; ++r) mt = fmaxf(mt, fmaxf(s0[r], s1[r]));
    mt = fmaxf(mt, __shfl_xor(mt, 16, 32));
    const float newM = fmaxf(M, mt);
    const float alpha = exp2f((M - newM) * cs);
    float rs = 0.0f;
#pragma unroll
    for (int r = 0; r < 8; ++r) {
      const float p0 = exp2f((s0[r] - newM) * cs);
      const float p1 = exp2f((s1[r] - newM) * cs);
      s0[r] = p0; s1[r] = p1; rs += p0 + p1;
    }
    rs += __shfl_xor(rs, 16, 32);
    L = L * alpha + rs;
    M = newM;
#pragma unroll
    for (int r = 0; r < 8; ++r) {
      acc0[r] *= alpha; acc1[r] *= alpha; acc2[r] *= alpha; acc3[r] *= alpha;
    }

    // ---- P^T D-frag -> B-frag: packs + one cross-half shfl ----
    v8u bp;
#pragma unroll
    for (int i = 0; i < 4; ++i) {
      const unsigned p1k = pk16(s0[2 * i], s0[2 * i + 1]);
      const unsigned p2k = pk16(s1[2 * i], s1[2 * i + 1]);
      const unsigned q1k = __shfl_xor(p1k, 16, 32);
      const unsigned q2k = __shfl_xor(p2k, 16, 32);
      bp[i]     = hf ? q2k : p1k;
      bp[4 + i] = hf ? p2k : q1k;
    }

    // ---- O^T += V^T * P^T ----
#pragma unroll
    for (int ch = 0; ch < 4; ++ch) {
      v8u av;
#pragma unroll
      for (int j = 0; j < 8; ++j) {
        av[j] = vlds32[vb + (ch * 16 + l16) * (VSTR / 2) + hf * 8 + j];
      }
      v8f& a = (ch == 0 ? acc0 : ch == 1 ? acc1 : ch == 2 ? acc2 : acc3);
      a = wmma_f16(av, bp, a);
    }
  }

  const float invL = 1.0f / L;
  float* op = of + (tokQ + l16) * CCH;
#pragma unroll
  for (int r = 0; r < 8; ++r) {
    const int c0 = r + 8 * hf;
    op[c0]      = acc0[r] * invL;
    op[16 + c0] = acc1[r] * invL;
    op[32 + c0] = acc2[r] * invL;
    op[48 + c0] = acc3[r] * invL;
  }
}

// ---------------------------------------------------------------------------
// Kernel 3: LN1(v+o) -> WMMA FFN -> LN2 -> [B,C,H,W].
// ---------------------------------------------------------------------------
__global__ __launch_bounds__(256) void epilogue_kernel(
    const float* __restrict__ vf, const float* __restrict__ of,
    const float* __restrict__ ln1w, const float* __restrict__ ln1b,
    const float* __restrict__ ln2w, const float* __restrict__ ln2b,
    const float* __restrict__ W1, const float* __restrict__ b1,
    const float* __restrict__ W2, const float* __restrict__ b2,
    float* __restrict__ out) {
  __shared__ _Float16 Wl[2][CCH * WSTR];
  {
    const float* Ws[2] = {W1, W2};
#pragma unroll
    for (int m = 0; m < 2; ++m) {
      for (int e = threadIdx.x; e < CCH * CCH; e += 256) {
        const int o = e >> 6, c = e & 63;
        Wl[m][o * WSTR + c] = (_Float16)Ws[m][e];
      }
    }
  }
  __syncthreads();

  const int lane = threadIdx.x & 31;
  const int wave = threadIdx.x >> 5;
  const int hf = lane >> 4, l16 = lane & 15;
  const int tok0 = blockIdx.x * 128 + wave * 16;
  const int b = tok0 >> 12;
  const int tok = tok0 + l16;
  const int n = tok & (NTOK - 1);

  // ---- x = v + o; LN1 (B layout: lane owns ch hf*16..+15 and +32..) ----
  const float* vp = vf + tok * CCH;
  const float* op = of + tok * CCH;
  float xl[16], xh[16];
  float s = 0.0f;
#pragma unroll
  for (int i = 0; i < 16; ++i) {
    const int c = hf * 16 + i;
    xl[i] = vp[c] + op[c];
    xh[i] = vp[c + 32] + op[c + 32];
    s += xl[i] + xh[i];
  }
  s += __shfl_xor(s, 16, 32);
  const float mu = s * (1.0f / CCH);
  float vv = 0.0f;
#pragma unroll
  for (int i = 0; i < 16; ++i) {
    float d = xl[i] - mu; vv = fmaf(d, d, vv);
    d = xh[i] - mu; vv = fmaf(d, d, vv);
  }
  vv += __shfl_xor(vv, 16, 32);
  const float rstd = rsqrtf(vv * (1.0f / CCH) + 1e-5f);
#pragma unroll
  for (int i = 0; i < 16; ++i) {
    const int c = hf * 16 + i;
    xl[i] = (xl[i] - mu) * rstd * ln1w[c] + ln1b[c];
    xh[i] = (xh[i] - mu) * rstd * ln1w[c + 32] + ln1b[c + 32];
  }

  v8u bxl, bxh;
#pragma unroll
  for (int j = 0; j < 8; ++j) {
    bxl[j] = pk16(xl[2 * j], xl[2 * j + 1]);
    bxh[j] = pk16(xh[2 * j], xh[2 * j + 1]);
  }

  const unsigned* w132 = (const unsigned*)&Wl[0][0];
  const unsigned* w232 = (const unsigned*)&Wl[1][0];

  // ---- h^T = relu(W1 x^T + b1), packed per chunk ----
  unsigned pkv[4][4];
#pragma unroll
  for (int chunk = 0; chunk < 4; ++chunk) {
    v8f d = {};
    d = wmma_f16(ldA(w132, chunk, 0, l16, hf), bxl, d);
    d = wmma_f16(ldA(w132, chunk, 1, l16, hf), bxh, d);
#pragma unroll
    for (int r = 0; r < 8; ++r) {
      const int ch = chunk * 16 + 8 * hf + r;
      d[r] = fmaxf(d[r] + b1[ch], 0.0f);
    }
#pragma unroll
    for (int i = 0; i < 4; ++i) pkv[chunk][i] = pk16(d[2 * i], d[2 * i + 1]);
  }
  v8u bhl, bhh;
#pragma unroll
  for (int i = 0; i < 4; ++i) {
    const unsigned q0 = __shfl_xor(pkv[0][i], 16, 32);
    const unsigned q1 = __shfl_xor(pkv[1][i], 16, 32);
    const unsigned q2 = __shfl_xor(pkv[2][i], 16, 32);
    const unsigned q3 = __shfl_xor(pkv[3][i], 16, 32);
    bhl[i]     = hf ? q1 : pkv[0][i];
    bhl[4 + i] = hf ? pkv[1][i] : q0;
    bhh[i]     = hf ? q3 : pkv[2][i];
    bhh[4 + i] = hf ? pkv[3][i] : q2;
  }

  // ---- residual x: B layout -> D layout (one f32 cross-half exchange) ----
  float ex1[8], ex2[8];
#pragma unroll
  for (int i = 0; i < 8; ++i) {
    ex1[i] = __shfl_xor(hf ? xl[i] : xl[8 + i], 16, 32);
    ex2[i] = __shfl_xor(hf ? xh[i] : xh[8 + i], 16, 32);
  }
  float xr[4][8];
#pragma unroll
  for (int r = 0; r < 8; ++r) {
    xr[0][r] = hf ? ex1[r] : xl[r];
    xr[1][r] = hf ? xl[8 + r] : ex1[r];
    xr[2][r] = hf ? ex2[r] : xh[r];
    xr[3][r] = hf ? xh[8 + r] : ex2[r];
  }

  // ---- ffn^T = W2 h^T + b2; residual; LN2 ----
  float y[4][8];
  float s2 = 0.0f;
#pragma unroll
  for (int chunk = 0; chunk < 4; ++chunk) {
    v8f d = {};
    d = wmma_f16(ldA(w232, chunk, 0, l16, hf), bhl, d);
    d = wmma_f16(ldA(w232, chunk, 1, l16, hf), bhh, d);
#pragma unroll
    for (int r = 0; r < 8; ++r) {
      const int ch = chunk * 16 + 8 * hf + r;
      y[chunk][r] = xr[chunk][r] + d[r] + b2[ch];
      s2 += y[chunk][r];
    }
  }
  s2 += __shfl_xor(s2, 16, 32);
  const float mu2 = s2 * (1.0f / CCH);
  float vv2 = 0.0f;
#pragma unroll
  for (int chunk = 0; chunk < 4; ++chunk)
#pragma unroll
    for (int r = 0; r < 8; ++r) {
      const float d = y[chunk][r] - mu2;
      vv2 = fmaf(d, d, vv2);
    }
  vv2 += __shfl_xor(vv2, 16, 32);
  const float rstd2 = rsqrtf(vv2 * (1.0f / CCH) + 1e-5f);

#pragma unroll
  for (int chunk = 0; chunk < 4; ++chunk)
#pragma unroll
    for (int r = 0; r < 8; ++r) {
      const int ch = chunk * 16 + 8 * hf + r;
      out[(b * CCH + ch) * NTOK + n] =
          (y[chunk][r] - mu2) * rstd2 * ln2w[ch] + ln2b[ch];
    }
}

// ---------------------------------------------------------------------------
extern "C" void kernel_launch(void* const* d_in, const int* in_sizes, int n_in,
                              void* d_out, int out_size, void* d_ws, size_t ws_size,
                              hipStream_t stream) {
  const float* seg  = (const float*)d_in[0];
  const float* gau  = (const float*)d_in[1];
  const float* Wq   = (const float*)d_in[2];
  const float* bq   = (const float*)d_in[3];
  const float* Wk   = (const float*)d_in[4];
  const float* bk   = (const float*)d_in[5];
  const float* Wv   = (const float*)d_in[6];
  const float* bv   = (const float*)d_in[7];
  const float* ln1w = (const float*)d_in[8];
  const float* ln1b = (const float*)d_in[9];
  const float* ln2w = (const float*)d_in[10];
  const float* ln2b = (const float*)d_in[11];
  const float* W1   = (const float*)d_in[12];
  const float* b1   = (const float*)d_in[13];
  const float* W2   = (const float*)d_in[14];
  const float* b2   = (const float*)d_in[15];
  float* out = (float*)d_out;

  // Workspace layout (14 MiB total):
  char* ws = (char*)d_ws;
  _Float16* qf  = (_Float16*)(ws);                       // 2 MiB  q   f16
  _Float16* kf  = (_Float16*)(ws + (2u << 20));          // 2 MiB  k   f16
  _Float16* vtf = (_Float16*)(ws + (4u << 20));          // 2 MiB  v^T f16
  float*    vf  = (float*)   (ws + (6u << 20));          // 4 MiB  v   f32
  float*    of  = (float*)   (ws + (10u << 20));         // 4 MiB  attn out f32

  proj_kernel<<<BN / 128, 256, 0, stream>>>(seg, gau, Wq, bq, Wk, bk, Wv, bv,
                                            qf, kf, vf, vtf);
  flash_attn_kernel<<<4 * (NTOK / 128), 256, 0, stream>>>(qf, kf, vtf, of);
  epilogue_kernel<<<BN / 128, 256, 0, stream>>>(vf, of, ln1w, ln1b, ln2w, ln2b,
                                                W1, b1, W2, b2, out);
}
